// SACNet_25984552141425
// MI455X (gfx1250) — compile-verified, matrix operand order checked by reading, not run on hardware
//
#include <hip/hip_runtime.h>
#include <hip/hip_bf16.h>

// ---------------------------------------------------------------------------
// CDNA5 (gfx1250) WMMA helpers: bf16 16x16x32, fp32 accumulate
// ---------------------------------------------------------------------------
typedef __attribute__((ext_vector_type(16))) __bf16 v16bf;
typedef __attribute__((ext_vector_type(8)))  float  v8f;

union BF16Frag { v16bf v; __bf16 h[16]; unsigned short s[16]; uint4 q[2]; };

// Native fptrunc f32->bf16 (lowers to HW cvt on gfx1250), bit pattern out.
__device__ __forceinline__ unsigned short f2bf_bits(float x){
  __bf16 h = (__bf16)x;
  return __builtin_bit_cast(unsigned short, h);
}

// Element e of a 16-half A/B fragment maps to K index (per ISA 16-bit layout):
//  lanes 0-15 (half=0): e0..7 -> K0..7,   e8..15 -> K16..23
//  lanes16-31 (half=1): e0..7 -> K8..15,  e8..15 -> K24..31
// i.e. per lane: two contiguous 8-element chunks at K = half*8 and 16+half*8.
__device__ __forceinline__ int koff(int e, int half){
  return ((e < 8) ? 0 : 16) + half * 8 + (e & 7);
}

__device__ __forceinline__ v8f wmma_bf16(const BF16Frag& A, const BF16Frag& B, v8f c){
  return __builtin_amdgcn_wmma_f32_16x16x32_bf16(false, A.v, false, B.v, (short)0, c, false, false);
}

// Load a row-contiguous bf16 fragment: chunk0 = [base + half*8, +8),
// chunk1 = [base + 16 + half*8, +8) -> 2x b128 loads.
__device__ __forceinline__ void load_frag_row(BF16Frag& F, const unsigned short* row, int fh){
  F.q[0] = *(const uint4*)(row + fh * 8);
  F.q[1] = *(const uint4*)(row + 16 + fh * 8);
}

// ---------------------------------------------------------------------------
// Implicit-GEMM 3x3 conv (VALID, dilation d), N(out ch)=64, fused bias+ReLU.
// out is channel-major [64][OH*OW].  One wave -> one 16x16 output tile.
// Per-block LDS im2col offset table kills the inner-loop div/mod.
// ---------------------------------------------------------------------------
__global__ __launch_bounds__(256) void conv3x3_wmma(
    const float* __restrict__ in, const float* __restrict__ w,
    const float* __restrict__ bias, float* __restrict__ out,
    int C, int H, int W, int OH, int OW, int dil, int numTiles)
{
  __shared__ int offT[928];                   // K <= 927
  const int M   = OH * OW;
  const int K   = C * 9;
  const int K32 = (K + 31) & ~31;
  for (int k = threadIdx.x; k < K; k += blockDim.x){
    int ci = k / 9, r = k - ci * 9, ky = r / 3, kx = r - ky * 3;
    offT[k] = (ci * H + ky * dil) * W + kx * dil;
  }
  __syncthreads();

  int wid = (blockIdx.x * blockDim.x + threadIdx.x) >> 5;
  if (wid >= numTiles) return;                // wave-uniform: EXEC stays full
  int lane = threadIdx.x & 31;
  int fi = lane & 15, fh = lane >> 4;
  int tM = wid >> 2, tN = wid & 3;            // ntN = 64/16 = 4
  int mBase = tM * 16, nBase = tN * 16;

  int m = mBase + fi;                         // A row this lane gathers
  float mMask = (m < M) ? 1.f : 0.f;
  int mc = min(m, M - 1);
  int oy = mc / OW, ox = mc - oy * OW;
  const float* __restrict__ rowPtr = in + oy * W + ox;
  const float* __restrict__ wRow   = w + (nBase + fi) * K;   // B column (<64 always)

  v8f acc = {0.f,0.f,0.f,0.f,0.f,0.f,0.f,0.f};
  for (int k0 = 0; k0 < K32; k0 += 32){
    BF16Frag A, B;
#pragma unroll
    for (int e = 0; e < 16; ++e){
      int k  = k0 + koff(e, fh);
      int kc = min(k, K - 1);
      float kMask = (k < K) ? 1.f : 0.f;
      float av = rowPtr[offT[kc]];            // always in-bounds (clamped)
      float bv = wRow[kc];
      A.h[e] = (__bf16)(av * (kMask * mMask));
      B.h[e] = (__bf16)(bv * kMask);
    }
    acc = wmma_bf16(A, B, acc);
  }
  int cn = nBase + fi;
#pragma unroll
  for (int r = 0; r < 8; ++r){
    int cm = mBase + r + 8 * fh;
    if (cm < M){
      float v = fmaxf(acc[r] + bias[cn], 0.f);
      out[cn * M + cm] = v;
    }
  }
}

// ---------------------------------------------------------------------------
// Generic strided GEMM: C(m,n) = sum_k A[m*Arm+k*Ack] * B[k*Brk+n*Bcn]
// optional bias[n], ReLU, residual add (post-ReLU), fp32 or bf16 output.
// Clamped addresses + value masks: no EXEC branching in the K loop.
// ---------------------------------------------------------------------------
__global__ __launch_bounds__(256) void gemm_wmma(
    const float* __restrict__ A, const float* __restrict__ B,
    const float* __restrict__ bias, const float* __restrict__ res,
    float* __restrict__ Cf, unsigned short* __restrict__ Cb,
    int M, int N, int K,
    int Arm, int Ack, int Brk, int Bcn, int Crm, int Ccn,
    int relu, int numTiles)
{
  int wid = (blockIdx.x * blockDim.x + threadIdx.x) >> 5;
  if (wid >= numTiles) return;
  int lane = threadIdx.x & 31;
  int fi = lane & 15, fh = lane >> 4;
  int ntN = (N + 15) >> 4;
  int tM = wid / ntN, tN = wid - tM * ntN;
  int mBase = tM * 16, nBase = tN * 16;
  int m = mBase + fi, n = nBase + fi;
  float aMask = (m < M) ? 1.f : 0.f;
  float bMask = (n < N) ? 1.f : 0.f;
  int mc = min(m, M - 1), nc = min(n, N - 1);
  const float* __restrict__ Am = A + (size_t)mc * Arm;
  const float* __restrict__ Bn = B + (size_t)nc * Bcn;
  int K32 = (K + 31) & ~31;

  v8f acc = {0.f,0.f,0.f,0.f,0.f,0.f,0.f,0.f};
  for (int k0 = 0; k0 < K32; k0 += 32){
    BF16Frag Af, Bf;
#pragma unroll
    for (int e = 0; e < 16; ++e){
      int k  = k0 + koff(e, fh);
      int kc = min(k, K - 1);
      float kMask = (k < K) ? 1.f : 0.f;
      Af.h[e] = (__bf16)(Am[(size_t)kc * Ack] * (kMask * aMask));
      Bf.h[e] = (__bf16)(Bn[(size_t)kc * Brk] * (kMask * bMask));
    }
    acc = wmma_bf16(Af, Bf, acc);
  }
  int cn = nBase + fi;
  if (cn < N){
#pragma unroll
    for (int r = 0; r < 8; ++r){
      int cm = mBase + r + 8 * fh;
      if (cm < M){
        float v = acc[r] + (bias ? bias[cn] : 0.f);
        if (relu) v = fmaxf(v, 0.f);
        int addr = cm * Crm + cn * Ccn;
        if (res) v += res[addr];
        if (Cb) Cb[addr] = f2bf_bits(v); else Cf[addr] = v;
      }
    }
  }
}

// ---------------------------------------------------------------------------
// Flash-attention pass 1: row-wise online (max, sumexp) of S = a @ bmat
// a_bf [HW][32], b_bf [32][HW] (bf16 bits).  One wave per 16-row tile.
// ---------------------------------------------------------------------------
__global__ __launch_bounds__(256) void attn_rowstats(
    const unsigned short* __restrict__ aB, const unsigned short* __restrict__ bB,
    float* __restrict__ rowM, float* __restrict__ rowZ, int HW)
{
  int wid = (blockIdx.x * blockDim.x + threadIdx.x) >> 5;
  if (wid >= (HW >> 4)) return;
  int lane = threadIdx.x & 31;
  int fi = lane & 15, fh = lane >> 4;
  int iBase = wid * 16;

  BF16Frag Af;
  load_frag_row(Af, aB + (iBase + fi) * 32, fh);   // 2x global_load_b128

  float rm[8], rs[8];
#pragma unroll
  for (int r = 0; r < 8; ++r){ rm[r] = -3.0e38f; rs[r] = 0.f; }

  for (int j0 = 0; j0 < HW; j0 += 16){
    BF16Frag Bf;
#pragma unroll
    for (int e = 0; e < 16; ++e) Bf.s[e] = bB[koff(e, fh) * HW + j0 + fi];
    v8f z = {0.f,0.f,0.f,0.f,0.f,0.f,0.f,0.f};
    v8f s = wmma_bf16(Af, Bf, z);
#pragma unroll
    for (int r = 0; r < 8; ++r){
      float sv = s[r];
      if (sv > rm[r]){ rs[r] = rs[r] * __expf(rm[r] - sv) + 1.f; rm[r] = sv; }
      else            rs[r] += __expf(sv - rm[r]);
    }
  }
  // reduce across the 16 lanes (differing column index) sharing each row
#pragma unroll
  for (int r = 0; r < 8; ++r){
    for (int mask = 1; mask < 16; mask <<= 1){
      float om = __shfl_xor(rm[r], mask, 32);
      float os = __shfl_xor(rs[r], mask, 32);
      float nm = fmaxf(rm[r], om);
      rs[r] = rs[r] * __expf(rm[r] - nm) + os * __expf(om - nm);
      rm[r] = nm;
    }
  }
  if (fi == 0){
#pragma unroll
    for (int r = 0; r < 8; ++r){
      int i = iBase + r + 8 * fh;
      rowM[i] = rm[r];
      rowZ[i] = rs[r];
    }
  }
}

// ---------------------------------------------------------------------------
// Flash-attention pass 2: ctxOut[t,j] = sum_i g[t,i] * exp(s_ij - m_i)/Z_i
// Recompute S tiles with WMMA, exponentiate, relayout P via LDS ([n][k] so
// the B-fragment reload is 2x ds_load_b128), accumulate Out = G x P.
// ---------------------------------------------------------------------------
__global__ __launch_bounds__(256) void attn_apply(
    const unsigned short* __restrict__ aB, const unsigned short* __restrict__ bB,
    const unsigned short* __restrict__ gB, const float* __restrict__ rowM,
    const float* __restrict__ rowZ, float* __restrict__ ctxOut, int HW)
{
  __shared__ float sM[6400];
  __shared__ float sZ[6400];
  __shared__ unsigned short P[8][16][40];     // [wave][n][k], padded row (80B)

  for (int idx = threadIdx.x; idx < HW; idx += blockDim.x){
    sM[idx] = rowM[idx];
    sZ[idx] = rowZ[idx];
  }
  __syncthreads();

  int wv = threadIdx.x >> 5;
  int tile = blockIdx.x * 8 + wv;
  int lane = threadIdx.x & 31;
  int fi = lane & 15, fh = lane >> 4;
  int jBase = tile * 16;
  bool act = tile < (HW >> 4);                // wave-uniform

  BF16Frag Bj;
  if (act){
#pragma unroll
    for (int e = 0; e < 16; ++e) Bj.s[e] = bB[koff(e, fh) * HW + jBase + fi];
  }
  v8f C0 = {0.f,0.f,0.f,0.f,0.f,0.f,0.f,0.f};
  v8f C1 = {0.f,0.f,0.f,0.f,0.f,0.f,0.f,0.f};

  for (int i0 = 0; i0 < HW; i0 += 32){
    if (act){
      BF16Frag A0, A1;
      load_frag_row(A0, aB + (i0 + fi) * 32, fh);
      load_frag_row(A1, aB + (i0 + 16 + fi) * 32, fh);
      v8f z = {0.f,0.f,0.f,0.f,0.f,0.f,0.f,0.f};
      v8f s0 = wmma_bf16(A0, Bj, z);
      v8f s1 = wmma_bf16(A1, Bj, z);
#pragma unroll
      for (int r = 0; r < 8; ++r){
        int i  = i0 + r + 8 * fh;
        int i2 = i + 16;
        P[wv][fi][r + 8 * fh]      = f2bf_bits(__expf(s0[r] - sM[i])  / sZ[i]);
        P[wv][fi][16 + r + 8 * fh] = f2bf_bits(__expf(s1[r] - sM[i2]) / sZ[i2]);
      }
    }
    __syncthreads();
    if (act){
      BF16Frag Pf, Ga, Gb;
      Pf.q[0] = *(const uint4*)&P[wv][fi][fh * 8];        // ds_load_b128
      Pf.q[1] = *(const uint4*)&P[wv][fi][16 + fh * 8];   // ds_load_b128
      load_frag_row(Ga, gB + fi * HW + i0, fh);
      load_frag_row(Gb, gB + (16 + fi) * HW + i0, fh);
      C0 = wmma_bf16(Ga, Pf, C0);
      C1 = wmma_bf16(Gb, Pf, C1);
    }
    __syncthreads();
  }
  if (act){
#pragma unroll
    for (int r = 0; r < 8; ++r){
      int t = r + 8 * fh;
      ctxOut[t * HW + jBase + fi]        = C0[r];
      ctxOut[(16 + t) * HW + jBase + fi] = C1[r];
    }
  }
}

// ---------------------------------------------------------------------------
// Elementwise / small kernels
// ---------------------------------------------------------------------------
__global__ void avgpool2k(const float* __restrict__ in, float* __restrict__ out,
                          int C, int H, int W, int OH, int OW)
{
  int idx = blockIdx.x * blockDim.x + threadIdx.x;
  int tot = C * OH * OW;
  if (idx >= tot) return;
  int ox = idx % OW; int t = idx / OW; int oy = t % OH; int c = t / OH;
  const float* p = in + (c * H + 2 * oy) * W + 2 * ox;
  out[idx] = 0.25f * (p[0] + p[1] + p[W] + p[W + 1]);
}

__global__ void resize_ac(const float* __restrict__ in, float* __restrict__ out,
                          int C, int H, int W, int OH, int OW)
{
  int idx = blockIdx.x * blockDim.x + threadIdx.x;
  int tot = C * OH * OW;
  if (idx >= tot) return;
  int ox = idx % OW; int t = idx / OW; int oy = t % OH; int c = t / OH;
  float fy = (OH == 1) ? 0.f : oy * (float)(H - 1) / (float)(OH - 1);
  int y0 = (int)fy; int y1 = min(y0 + 1, H - 1); float wy = fy - (float)y0;
  float fx = (OW == 1) ? 0.f : ox * (float)(W - 1) / (float)(OW - 1);
  int x0 = (int)fx; int x1 = min(x0 + 1, W - 1); float wx = fx - (float)x0;
  const float* b = in + c * H * W;
  float top = b[y0 * W + x0] * (1.f - wx) + b[y0 * W + x1] * wx;
  float bot = b[y1 * W + x0] * (1.f - wx) + b[y1 * W + x1] * wx;
  out[idx] = top * (1.f - wy) + bot * wy;
}

__global__ void vq_prep(const float* __restrict__ cw, float* __restrict__ C2)
{
  int k = threadIdx.x;
  if (k < 48){
    float s = 0.f;
#pragma unroll
    for (int d = 0; d < 32; ++d){ float v = cw[k * 32 + d]; s += v * v; }
    C2[k] = s;
  }
}

__global__ __launch_bounds__(128) void vq_assign(
    const float* __restrict__ Z, const float* __restrict__ cw,
    const float* __restrict__ scale, const float* __restrict__ C2,
    float* __restrict__ A, int Np)
{
  int n = blockIdx.x * blockDim.x + threadIdx.x;
  if (n >= Np) return;
  float z[32]; float x2 = 0.f;
#pragma unroll
  for (int d = 0; d < 32; ++d){ z[d] = Z[n * 32 + d]; x2 += z[d] * z[d]; }
  float dist[48]; float mx = -3.0e38f;
  for (int k = 0; k < 48; ++k){
    float dot = 0.f;
#pragma unroll
    for (int d = 0; d < 32; ++d) dot += z[d] * cw[k * 32 + d];
    float dv = scale[k] * (x2 + C2[k] - 2.f * dot);
    dist[k] = dv; mx = fmaxf(mx, dv);
  }
  float s = 0.f;
  for (int k = 0; k < 48; ++k){ float e = __expf(dist[k] - mx); dist[k] = e; s += e; }
  float inv = 1.f / s;
  for (int k = 0; k < 48; ++k) A[n * 48 + k] = dist[k] * inv;
}

__global__ void colsum48(const float* __restrict__ A, float* __restrict__ out, int Np)
{
  __shared__ float red[256];
  int k = blockIdx.x;
  float s = 0.f;
  for (int n = threadIdx.x; n < Np; n += blockDim.x) s += A[n * 48 + k];
  red[threadIdx.x] = s;
  __syncthreads();
  for (int off = 128; off; off >>= 1){
    if (threadIdx.x < off) red[threadIdx.x] += red[threadIdx.x + off];
    __syncthreads();
  }
  if (threadIdx.x == 0) out[k] = red[0];
}

__global__ void vq_gate(float* __restrict__ E, const float* __restrict__ Asum,
                        const float* __restrict__ cw, const float* __restrict__ bnw,
                        const float* __restrict__ bnb, const float* __restrict__ attw,
                        const float* __restrict__ attb, float* __restrict__ gate)
{
  __shared__ float S[48][32];
  __shared__ float Esum[32];
  int t = threadIdx.x;
  if (t < 48){
    float row[32]; float mean = 0.f;
#pragma unroll
    for (int d = 0; d < 32; ++d){
      float v = E[t * 32 + d] - Asum[t] * cw[t * 32 + d];
      row[d] = v; mean += v;
    }
    mean *= (1.f / 32.f);
    float var = 0.f;
#pragma unroll
    for (int d = 0; d < 32; ++d){ float dv = row[d] - mean; var += dv * dv; }
    var *= (1.f / 32.f);
    float inv = rsqrtf(var + 1e-5f);
#pragma unroll
    for (int d = 0; d < 32; ++d){
      float v = (row[d] - mean) * inv * bnw[t] + bnb[t];
      S[t][d] = fmaxf(v, 0.f);
    }
  }
  __syncthreads();
  if (t < 32){
    float s = 0.f;
    for (int k = 0; k < 48; ++k) s += S[k][t];
    Esum[t] = s;
  }
  __syncthreads();
  if (t < 64){
    float g = attb[t];
#pragma unroll
    for (int d = 0; d < 32; ++d) g += attw[t * 32 + d] * Esum[d];
    gate[t] = 1.f / (1.f + __expf(-g));
  }
}

__global__ void gate_apply(float* __restrict__ x, const float* __restrict__ gate, int M)
{
  int idx = blockIdx.x * blockDim.x + threadIdx.x;
  if (idx >= 64 * M) return;
  int c = idx / M;
  x[idx] *= (1.f + gate[c]);
}

// Fused tri-source bilinear upsample + concat + 1x1 classifier conv.
__device__ __forceinline__ void cls_accum(const float* __restrict__ src, int H, int W,
                                          int oy, int ox, const float* __restrict__ wb,
                                          float acc[9])
{
  float fy = oy * (float)(H - 1) / 339.f;
  int y0 = (int)fy; int y1 = min(y0 + 1, H - 1); float wy = fy - (float)y0;
  float fx = ox * (float)(W - 1) / 339.f;
  int x0 = (int)fx; int x1 = min(x0 + 1, W - 1); float wx = fx - (float)x0;
  for (int c = 0; c < 64; ++c){
    const float* pl = src + c * H * W;
    float top = pl[y0 * W + x0] * (1.f - wx) + pl[y0 * W + x1] * wx;
    float bot = pl[y1 * W + x0] * (1.f - wx) + pl[y1 * W + x1] * wx;
    float v = top * (1.f - wy) + bot * wy;
#pragma unroll
    for (int nc = 0; nc < 9; ++nc) acc[nc] += wb[nc * 192 + c] * v;
  }
}

__global__ __launch_bounds__(128) void classifier(
    const float* __restrict__ c1, const float* __restrict__ c2,
    const float* __restrict__ c3, const float* __restrict__ w,
    const float* __restrict__ b, float* __restrict__ out)
{
  const int OH = 340, OW = 340;
  int p = blockIdx.x * blockDim.x + threadIdx.x;
  if (p >= OH * OW) return;
  int oy = p / OW, ox = p - oy * OW;
  float acc[9];
#pragma unroll
  for (int nc = 0; nc < 9; ++nc) acc[nc] = b[nc];
  cls_accum(c1, 338, 338, oy, ox, w,        acc);
  cls_accum(c2, 334, 334, oy, ox, w + 64,   acc);
  cls_accum(c3, 161, 161, oy, ox, w + 128,  acc);
#pragma unroll
  for (int nc = 0; nc < 9; ++nc) out[nc * OH * OW + p] = acc[nc];
}

// ---------------------------------------------------------------------------
// Host launch
// ---------------------------------------------------------------------------
static inline int cdiv(int a, int b){ return (a + b - 1) / b; }

extern "C" void kernel_launch(void* const* d_in, const int* in_sizes, int n_in,
                              void* d_out, int out_size, void* d_ws, size_t ws_size,
                              hipStream_t stream)
{
  const float* x     = (const float*)d_in[0];
  const float* c0w   = (const float*)d_in[1];
  const float* c0b   = (const float*)d_in[2];
  const float* c1w   = (const float*)d_in[3];
  const float* c1b   = (const float*)d_in[4];
  const float* c2w   = (const float*)d_in[5];
  const float* c2b   = (const float*)d_in[6];
  const float* alw   = (const float*)d_in[7];
  const float* bew   = (const float*)d_in[8];
  const float* gaw   = (const float*)d_in[9];
  const float* dew   = (const float*)d_in[10];
  const float* enw   = (const float*)d_in[11];
  const float* cwds  = (const float*)d_in[12];
  const float* scale = (const float*)d_in[13];
  const float* attw  = (const float*)d_in[14];
  const float* attb  = (const float*)d_in[15];
  const float* bnw   = (const float*)d_in[16];
  const float* bnb   = (const float*)d_in[17];
  const float* clsw  = (const float*)d_in[18];
  const float* clsb  = (const float*)d_in[19];
  float* out = (float*)d_out;

  char* W = (char*)d_ws;
  size_t off = 0;
  auto alloc = [&](size_t bytes)->char*{
    char* p = W + off;
    off = (off + bytes + 255) & ~(size_t)255;
    return p;
  };
  const int M0 = 338*338, M1 = 334*334, MP = 167*167, M3 = 161*161, HW = 6400;
  float* y0    = (float*)alloc((size_t)64*M0*4);   // conv1f
  float* y1    = (float*)alloc((size_t)64*M1*4);   // conv2f
  float* p1    = (float*)alloc((size_t)64*MP*4);
  float* x3    = (float*)alloc((size_t)64*M3*4);
  float* xh    = (float*)alloc((size_t)64*HW*4);
  float* x3b   = (float*)alloc((size_t)64*M3*4);   // deta-out (+res, later gated)
  unsigned short* aB = (unsigned short*)alloc((size_t)HW*32*2);
  unsigned short* bB = (unsigned short*)alloc((size_t)32*HW*2);
  unsigned short* gB = (unsigned short*)alloc((size_t)32*HW*2);
  float* rowM  = (float*)alloc((size_t)HW*4);
  float* rowZ  = (float*)alloc((size_t)HW*4);
  float* ctxO  = (float*)alloc((size_t)32*HW*4);
  float* ctxU  = (float*)alloc((size_t)32*M3*4);
  float* Zf    = (float*)alloc((size_t)M3*32*4);
  float* Abuf  = (float*)alloc((size_t)M3*48*4);
  float* Ebuf  = (float*)alloc((size_t)48*32*4);
  float* Asum  = (float*)alloc(48*4);
  float* C2    = (float*)alloc(48*4);
  float* gate  = (float*)alloc(64*4);
  (void)ws_size; (void)n_in; (void)in_sizes; (void)out_size;

  // conv0: 103 -> 64, dil 1
  { int t = cdiv(M0,16)*4;
    conv3x3_wmma<<<cdiv(t,8),256,0,stream>>>(x, c0w, c0b, y0, 103,340,340,338,338,1,t); }
  // conv1: 64 -> 64, dil 2
  { int t = cdiv(M1,16)*4;
    conv3x3_wmma<<<cdiv(t,8),256,0,stream>>>(y0, c1w, c1b, y1, 64,338,338,334,334,2,t); }
  // pool 334 -> 167
  { int tot = 64*MP; avgpool2k<<<cdiv(tot,256),256,0,stream>>>(y1, p1, 64,334,334,167,167); }
  // conv2: 64 -> 64, dil 3
  { int t = cdiv(M3,16)*4;
    conv3x3_wmma<<<cdiv(t,8),256,0,stream>>>(p1, c2w, c2b, x3, 64,167,167,161,161,3,t); }
  // pool 161 -> 80
  { int tot = 64*HW; avgpool2k<<<cdiv(tot,256),256,0,stream>>>(x3, xh, 64,161,161,80,80); }

  // alpha/beta/gamma 1x1 convs -> bf16 operands for the attention passes
  { int t = cdiv(HW,16)*2;
    gemm_wmma<<<cdiv(t,8),256,0,stream>>>(xh, alw, nullptr, nullptr, nullptr, aB,
        HW,32,64, 1,HW, 1,64, 32,1, 0, t);                 // a  [hw][t]
    gemm_wmma<<<cdiv(t,8),256,0,stream>>>(xh, bew, nullptr, nullptr, nullptr, bB,
        HW,32,64, 1,HW, 1,64, 1,HW, 0, t);                 // b  [t][hw]
    gemm_wmma<<<cdiv(t,8),256,0,stream>>>(xh, gaw, nullptr, nullptr, nullptr, gB,
        HW,32,64, 1,HW, 1,64, 1,HW, 1, t);                 // g  [t][hw] (ReLU)
  }

  // flash-style attention (no 6400x6400 materialization)
  attn_rowstats<<<50,256,0,stream>>>(aB, bB, rowM, rowZ, HW);
  attn_apply  <<<50,256,0,stream>>>(aB, bB, gB, rowM, rowZ, ctxO, HW);

  // upsample ctx 80 -> 161 (align_corners)
  { int tot = 32*M3; resize_ac<<<cdiv(tot,256),256,0,stream>>>(ctxO, ctxU, 32,80,80,161,161); }

  // deta 1x1 conv + ReLU + residual(x3)
  { int t = cdiv(M3,16)*4;
    gemm_wmma<<<cdiv(t,8),256,0,stream>>>(ctxU, dew, nullptr, x3, x3b, nullptr,
        M3,64,32, 1,M3, 1,32, 1,M3, 1, t); }

  // enc 1x1 conv + ReLU -> Z [N][32]
  { int t = cdiv(M3,16)*2;
    gemm_wmma<<<cdiv(t,8),256,0,stream>>>(x3b, enw, nullptr, nullptr, Zf, nullptr,
        M3,32,64, 1,M3, 1,64, 32,1, 1, t); }

  // soft-VQ
  vq_prep  <<<1,64,0,stream>>>(cwds, C2);
  vq_assign<<<cdiv(M3,128),128,0,stream>>>(Zf, cwds, scale, C2, Abuf, M3);
  colsum48 <<<48,256,0,stream>>>(Abuf, Asum, M3);
  { int t = 3*2;  // E = A^T Z  : [48 x 32], K = 25921
    gemm_wmma<<<1,256,0,stream>>>(Abuf, Zf, nullptr, nullptr, Ebuf, nullptr,
        48,32,M3, 1,48, 32,1, 32,1, 0, t); }
  vq_gate<<<1,64,0,stream>>>(Ebuf, Asum, cwds, bnw, bnb, attw, attb, gate);
  { int tot = 64*M3; gate_apply<<<cdiv(tot,256),256,0,stream>>>(x3b, gate, M3); }

  // fused upsample + concat + 1x1 classifier
  classifier<<<cdiv(340*340,128),128,0,stream>>>(y0, y1, x3b, clsw, clsb, out);
}
